// GemmaAttention_13297218748444
// MI455X (gfx1250) — matrix operand order repaired and checked by reading, not compiled, not run on hardware
//
#include <hip/hip_runtime.h>
#include <cstdint>

typedef __attribute__((ext_vector_type(16))) __bf16 v16bf;
typedef __attribute__((ext_vector_type(8)))  float  v8f;

union FragBF {
  uint4 u[2];
  v16bf v;
};

__device__ __forceinline__ uint16_t f2bfu(float f) {
  union { float f; uint32_t u; } v;
  v.f = f;
  uint32_t u = v.u;
  u += 0x7FFFu + ((u >> 16) & 1u);   // round-to-nearest-even bf16
  return (uint16_t)(u >> 16);
}

// ---- CDNA5 async global->LDS copy (ASYNCcnt-tracked, no VGPR round trip) ----
__device__ __forceinline__ void async_b128(uint32_t lds_off, const void* gaddr) {
  asm volatile("global_load_async_to_lds_b128 %0, %1, off"
               :: "v"(lds_off), "v"(gaddr)
               : "memory");
}
__device__ __forceinline__ void wait_async0() {
  asm volatile("s_wait_asynccnt 0x0" ::: "memory");
}
__device__ __forceinline__ uint32_t lds_off_of(const void* p) {
  return (uint32_t)(uintptr_t)p;     // generic LDS addr: low 32 bits = LDS byte offset
}

// A-fragment (16x32 bf16, row-major in LDS): lane l holds row (l&15);
// K elems {kb..kb+7, 16+kb..16+kb+7}, kb = (l>=16)?8:0.
__device__ __forceinline__ v16bf frag_rowmajor(const uint16_t* base, int row,
                                               int stride_elems, int k0, int lane) {
  const int kb = (lane & 16) ? 8 : 0;
  const char* p = (const char*)base + (size_t)row * stride_elems * 2u + (size_t)(k0 + kb) * 2u;
  FragBF f;
  f.u[0] = *(const uint4*)p;
  f.u[1] = *(const uint4*)(p + 32);
  return f.v;
}

// B-fragment (32x16 bf16) from K-contiguous [n][k] LDS layout:
// lane l holds col (l&15); K elems {kb..kb+7, 16+kb..16+kb+7}, kb = (l>=16)?8:0.
__device__ __forceinline__ v16bf frag_kcontig(const uint16_t* base, int n,
                                              int stride_elems, int k0, int lane) {
  const int kb = (lane & 16) ? 8 : 0;
  const char* p = (const char*)base + (size_t)n * stride_elems * 2u + (size_t)(k0 + kb) * 2u;
  FragBF f;
  f.u[0] = *(const uint4*)p;
  f.u[1] = *(const uint4*)(p + 32);
  return f.v;
}

__device__ __forceinline__ v8f wmma_bf16(v16bf a, v16bf b, v8f c) {
  return __builtin_amdgcn_wmma_f32_16x16x32_bf16(false, a, false, b, (short)0, c, false, false);
}

// ---------------------------------------------------------------------------
// One-time f32 -> packed bf16 convert (pairs -> u32 stores).
// ---------------------------------------------------------------------------
__global__ __launch_bounds__(256) void convert_pack(const float* __restrict__ src,
                                                    uint16_t* __restrict__ dst, int n2) {
  int i = blockIdx.x * 256 + threadIdx.x;
  if (i >= n2) return;
  float2 v = ((const float2*)src)[i];
  uint32_t p = (uint32_t)f2bfu(v.x) | ((uint32_t)f2bfu(v.y) << 16);
  ((uint32_t*)dst)[i] = p;
}

// ---------------------------------------------------------------------------
// One-time transpose + convert: B[K][N] f32 -> BT[N][K] bf16 (32x32 LDS tiles).
// ---------------------------------------------------------------------------
__global__ __launch_bounds__(256) void transpose_convert(const float* __restrict__ B,
                                                         uint16_t* __restrict__ BT,
                                                         int K, int N) {
  __shared__ float tile[32][33];
  const int n0 = blockIdx.x * 32;
  const int k0 = blockIdx.y * 32;
  const int tx = threadIdx.x & 31;
  const int ty = threadIdx.x >> 5;   // 0..7
#pragma unroll
  for (int i = 0; i < 4; ++i) {
    int k = ty + i * 8;
    tile[k][tx] = B[(size_t)(k0 + k) * N + n0 + tx];
  }
  __syncthreads();
#pragma unroll
  for (int i = 0; i < 4; ++i) {
    int n = ty + i * 8;
    BT[(size_t)(n0 + n) * K + k0 + tx] = f2bfu(tile[tx][n]);
  }
}

// ---------------------------------------------------------------------------
// C[M,N] f32 = A[M,K] bf16 * BT[N,K] bf16.  Block tile 128x128, 8 waves (2x4),
// wave tile 64x32.  Double-buffered LDS tiles filled by async b128 copies.
// ---------------------------------------------------------------------------
__global__ __launch_bounds__(256) void gemm_bf16_async(
    const uint16_t* __restrict__ A, const uint16_t* __restrict__ BT,
    float* __restrict__ C, int M, int N, int K) {
  __shared__ __align__(16) uint16_t Asl[2][128 * 32];   // [m][k]
  __shared__ __align__(16) uint16_t Bsl[2][128 * 32];   // [n][k]
  const int tid  = threadIdx.x;
  const int lane = tid & 31;
  const int wave = tid >> 5;
  const int cn   = lane & 15;
  const int bm = blockIdx.y * 128;
  const int bn = blockIdx.x * 128;
  const int wm = (wave >> 2) * 64;
  const int wn = (wave & 3) * 32;

  v8f acc[4][2];
#pragma unroll
  for (int i = 0; i < 4; ++i)
#pragma unroll
    for (int j = 0; j < 2; ++j) acc[i][j] = 0.0f;

  // Each tile is 128 rows x 64B; 512 chunks of 16B; 2 chunks/thread per matrix.
  auto copy_tile = [&](int k0, int b) {
#pragma unroll
    for (int i = 0; i < 2; ++i) {
      int c   = tid + i * 256;        // 0..511
      int row = c >> 2;               // 0..127
      int off = (c & 3) * 16;         // 0..48
      async_b128(lds_off_of(&Asl[b][row * 32]) + off,
                 (const char*)A + ((size_t)(bm + row) * K + k0) * 2 + off);
      async_b128(lds_off_of(&Bsl[b][row * 32]) + off,
                 (const char*)BT + ((size_t)(bn + row) * K + k0) * 2 + off);
    }
  };

  copy_tile(0, 0);
  wait_async0();
  __syncthreads();

  const int nT = K / 32;
  for (int t = 0; t < nT; ++t) {
    if (t + 1 < nT) copy_tile((t + 1) * 32, (t + 1) & 1);
    const int b = t & 1;
    v16bf bf0 = frag_kcontig(&Bsl[b][0], wn + 0  + cn, 32, 0, lane);
    v16bf bf1 = frag_kcontig(&Bsl[b][0], wn + 16 + cn, 32, 0, lane);
#pragma unroll
    for (int ms = 0; ms < 4; ++ms) {
      v16bf af = frag_rowmajor(&Asl[b][0], wm + ms * 16 + cn, 32, 0, lane);
      acc[ms][0] = wmma_bf16(af, bf0, acc[ms][0]);
      acc[ms][1] = wmma_bf16(af, bf1, acc[ms][1]);
    }
    wait_async0();      // next tile's async copies complete (per wave)
    __syncthreads();    // all waves' copies + reads complete
  }

  const int hl = (lane >> 4) & 1;
#pragma unroll
  for (int ms = 0; ms < 4; ++ms)
#pragma unroll
    for (int ns = 0; ns < 2; ++ns)
#pragma unroll
      for (int r = 0; r < 8; ++r) {
        int row = bm + wm + ms * 16 + hl * 8 + r;
        int col = bn + wn + ns * 16 + cn;
        C[(size_t)row * N + col] = acc[ms][ns][r];
      }
}

// ---------------------------------------------------------------------------
// RoPE (f32 in) fused with bf16 convert for Q [4096][8][256] and K [4096][256].
// ---------------------------------------------------------------------------
__global__ __launch_bounds__(256) void rope_convert(
    const float* __restrict__ Qf, const float* __restrict__ Kf,
    const int* __restrict__ pos_ids,
    uint16_t* __restrict__ Qb, uint16_t* __restrict__ Kb) {
  int idx = blockIdx.x * blockDim.x + threadIdx.x;
  const int total = 4096 * 9 * 128;     // 8 Q heads + 1 K head, 128 pairs each
  if (idx >= total) return;
  int d  = idx & 127;
  int t  = idx >> 7;
  int hh = t % 9;
  int s  = t / 9;
  float pos  = (float)pos_ids[s];
  float freq = __powf(10000.0f, -(float)(2 * d) * (1.0f / 256.0f));
  float ang  = pos * freq;
  float c  = __cosf(ang);
  float sn = __sinf(ang);
  const float* src;
  uint16_t* dst;
  if (hh < 8) { src = Qf + (size_t)s * 2048 + hh * 256; dst = Qb + (size_t)s * 2048 + hh * 256; }
  else        { src = Kf + (size_t)s * 256;             dst = Kb + (size_t)s * 256; }
  float x1 = src[d];
  float x2 = src[d + 128];
  dst[d]       = f2bfu(x1 * c - x2 * sn);
  dst[d + 128] = f2bfu(x2 * c + x1 * sn);
}

// ---------------------------------------------------------------------------
// Flash attention: block = (head, 128 q-rows), 8 waves x 16 rows.
// Q/K/V tiles are bf16 in global; K and V^T tiles double-buffered in LDS via
// async b128 copies.  Online softmax f32; QK^T and P*V on WMMA bf16.
// Output written bf16 to AOb [4096][2048] (A operand of the O projection).
// ---------------------------------------------------------------------------
__global__ __launch_bounds__(256) void flash_attn(
    const uint16_t* __restrict__ Qb, const uint16_t* __restrict__ Kb,
    const uint16_t* __restrict__ VTb, const float* __restrict__ mask,
    uint16_t* __restrict__ AOb) {
  constexpr int S = 4096;
  __shared__ __align__(16) uint16_t Qs[128 * 256];      // 64 KB  [qrow][d]
  __shared__ __align__(16) uint16_t Ks[2][32 * 256];    // 32 KB  [key][d]
  __shared__ __align__(16) uint16_t VsT[2][256 * 32];   // 32 KB  [d][key]
  __shared__ __align__(16) uint16_t Ps[8][16 * 32];     //  8 KB  per-wave P scratch

  const int tid  = threadIdx.x;
  const int lane = tid & 31;
  const int wave = tid >> 5;
  const int h  = blockIdx.y;
  const int q0 = blockIdx.x * 128;
  const int hl = (lane >> 4) & 1;
  const int cn = lane & 15;
  const int qr = wave * 16;

  auto copy_kv = [&](int kt2, int b) {
    const int kbase2 = kt2 * 32;
    // K tile: 32 rows x 512B = 1024 chunks
#pragma unroll
    for (int i = 0; i < 4; ++i) {
      int c    = tid + i * 256;
      int krow = c >> 5;              // 0..31
      int off  = (c & 31) * 16;       // 0..496
      async_b128(lds_off_of(&Ks[b][krow * 256]) + off,
                 (const char*)Kb + ((size_t)(kbase2 + krow) * 256) * 2 + off);
    }
    // V^T tile: 256 rows x 64B = 1024 chunks
#pragma unroll
    for (int i = 0; i < 4; ++i) {
      int c    = tid + i * 256;
      int drow = c >> 2;              // 0..255
      int off  = (c & 3) * 16;        // 0..48
      async_b128(lds_off_of(&VsT[b][drow * 32]) + off,
                 (const char*)VTb + ((size_t)drow * S + kbase2) * 2 + off);
    }
  };

  // Q tile: 128 rows x 512B = 4096 chunks
#pragma unroll
  for (int i = 0; i < 16; ++i) {
    int c   = tid + i * 256;
    int r   = c >> 5;                 // 0..127
    int off = (c & 31) * 16;
    async_b128(lds_off_of(&Qs[r * 256]) + off,
               (const char*)Qb + ((size_t)(q0 + r) * 2048 + h * 256) * 2 + off);
  }
  copy_kv(0, 0);
  wait_async0();
  __syncthreads();

  v8f O[16];
#pragma unroll
  for (int t = 0; t < 16; ++t) O[t] = 0.0f;
  float mi[8], li[8];
#pragma unroll
  for (int r = 0; r < 8; ++r) { mi[r] = -1e30f; li[r] = 0.0f; }

  for (int kt = 0; kt < S / 32; ++kt) {
    if (kt + 1 < S / 32) copy_kv(kt + 1, (kt + 1) & 1);
    const int b = kt & 1;
    const int kbase = kt * 32;

    // S tile: 16 q-rows x 32 keys, accumulate over D=256 in 8 WMMA steps.
    float sv[2][8];
#pragma unroll
    for (int ns = 0; ns < 2; ++ns) {
      v8f a = 0.0f;
#pragma unroll
      for (int dk = 0; dk < 8; ++dk) {
        v16bf af  = frag_rowmajor(Qs, qr + cn, 256, dk * 32, lane);
        v16bf bfv = frag_kcontig(&Ks[b][0], ns * 16 + cn, 256, dk * 32, lane);
        a = wmma_bf16(af, bfv, a);
      }
#pragma unroll
      for (int r = 0; r < 8; ++r) {
        int row = q0 + qr + hl * 8 + r;
        int col = kbase + ns * 16 + cn;
        sv[ns][r] = a[r] * 0.0625f + mask[(size_t)row * S + col];
      }
    }

    // Online softmax: rows live in 16-lane halves -> xor-reduce over 1,2,4,8.
    float alph[8];
#pragma unroll
    for (int r = 0; r < 8; ++r) {
      float rmx = fmaxf(sv[0][r], sv[1][r]);
      rmx = fmaxf(rmx, __shfl_xor(rmx, 1, 32));
      rmx = fmaxf(rmx, __shfl_xor(rmx, 2, 32));
      rmx = fmaxf(rmx, __shfl_xor(rmx, 4, 32));
      rmx = fmaxf(rmx, __shfl_xor(rmx, 8, 32));
      float mn = fmaxf(mi[r], rmx);
      float al = __expf(mi[r] - mn);
      float p0 = __expf(sv[0][r] - mn);
      float p1 = __expf(sv[1][r] - mn);
      float ps = p0 + p1;
      ps += __shfl_xor(ps, 1, 32);
      ps += __shfl_xor(ps, 2, 32);
      ps += __shfl_xor(ps, 4, 32);
      ps += __shfl_xor(ps, 8, 32);
      li[r] = li[r] * al + ps;
      mi[r] = mn;
      sv[0][r] = p0;
      sv[1][r] = p1;
      alph[r] = al;
    }
#pragma unroll
    for (int t = 0; t < 16; ++t)
#pragma unroll
      for (int r = 0; r < 8; ++r) O[t][r] *= alph[r];

    // P: C-layout regs -> LDS -> A-fragment (same-wave LDS ops are in-order).
    uint16_t* pw = &Ps[wave][0];
#pragma unroll
    for (int ns = 0; ns < 2; ++ns)
#pragma unroll
      for (int r = 0; r < 8; ++r)
        pw[(hl * 8 + r) * 32 + ns * 16 + cn] = f2bfu(sv[ns][r]);

    v16bf pf = frag_rowmajor(pw, cn, 32, 0, lane);
#pragma unroll
    for (int t = 0; t < 16; ++t) {
      v16bf vf = frag_kcontig(&VsT[b][0], t * 16 + cn, 32, 0, lane);
      O[t] = wmma_bf16(pf, vf, O[t]);
    }
    wait_async0();
    __syncthreads();
  }

#pragma unroll
  for (int t = 0; t < 16; ++t)
#pragma unroll
    for (int r = 0; r < 8; ++r) {
      int row = q0 + qr + hl * 8 + r;
      int col = h * 256 + t * 16 + cn;
      AOb[(size_t)row * 2048 + col] = f2bfu(O[t][r] / li[r]);
    }
}

// ---------------------------------------------------------------------------
extern "C" void kernel_launch(void* const* d_in, const int* in_sizes, int n_in,
                              void* d_out, int out_size, void* d_ws, size_t ws_size,
                              hipStream_t stream) {
  const float* hs   = (const float*)d_in[0];
  const float* mask = (const float*)d_in[1];
  const int*   pos  = (const int*)d_in[2];
  const float* Wq   = (const float*)d_in[3];
  const float* Wk   = (const float*)d_in[4];
  const float* Wv   = (const float*)d_in[5];
  const float* Wo   = (const float*)d_in[6];
  float* out = (float*)d_out;

  // Workspace layout (f32 region then bf16 region).
  char* w = (char*)d_ws;
  float* Qf = (float*)w;                 w += (size_t)4096 * 2048 * 4;   // Q f32
  float* Kf = (float*)w;                 w += (size_t)4096 * 256 * 4;    // K f32
  float* Vf = (float*)w;                 w += (size_t)4096 * 256 * 4;    // V f32
  uint16_t* hsb  = (uint16_t*)w;         w += (size_t)4096 * 2048 * 2;   // hs bf16
  uint16_t* WqT  = (uint16_t*)w;         w += (size_t)2048 * 2048 * 2;   // Wq^T bf16
  uint16_t* WkT  = (uint16_t*)w;         w += (size_t)256  * 2048 * 2;
  uint16_t* WvT  = (uint16_t*)w;         w += (size_t)256  * 2048 * 2;
  uint16_t* WoT  = (uint16_t*)w;         w += (size_t)2048 * 2048 * 2;
  uint16_t* Qb   = (uint16_t*)w;         w += (size_t)4096 * 2048 * 2;   // roped Q bf16
  uint16_t* Kb   = (uint16_t*)w;         w += (size_t)4096 * 256 * 2;    // roped K bf16
  uint16_t* VTb  = (uint16_t*)w;         w += (size_t)256  * 4096 * 2;   // V^T bf16
  uint16_t* AOb  = (uint16_t*)w;         w += (size_t)4096 * 2048 * 2;   // attn out bf16

  dim3 blk(256);

  // One-time bf16 conversions / transposes.
  convert_pack<<<(4096 * 2048 / 2 + 255) / 256, blk, 0, stream>>>(hs, hsb, 4096 * 2048 / 2);
  transpose_convert<<<dim3(2048 / 32, 2048 / 32), blk, 0, stream>>>(Wq, WqT, 2048, 2048);
  transpose_convert<<<dim3(256 / 32, 2048 / 32),  blk, 0, stream>>>(Wk, WkT, 2048, 256);
  transpose_convert<<<dim3(256 / 32, 2048 / 32),  blk, 0, stream>>>(Wv, WvT, 2048, 256);
  transpose_convert<<<dim3(2048 / 32, 2048 / 32), blk, 0, stream>>>(Wo, WoT, 2048, 2048);

  // QKV projections (bf16 WMMA, async-staged tiles), f32 outputs.
  gemm_bf16_async<<<dim3(2048 / 128, 4096 / 128), blk, 0, stream>>>(hsb, WqT, Qf, 4096, 2048, 2048);
  gemm_bf16_async<<<dim3(256 / 128, 4096 / 128),  blk, 0, stream>>>(hsb, WkT, Kf, 4096, 256, 2048);
  gemm_bf16_async<<<dim3(256 / 128, 4096 / 128),  blk, 0, stream>>>(hsb, WvT, Vf, 4096, 256, 2048);

  // RoPE (f32) fused with bf16 convert; V transposed+converted for P*V B-frags.
  const int rope_total = 4096 * 9 * 128;
  rope_convert<<<(rope_total + 255) / 256, blk, 0, stream>>>(Qf, Kf, pos, Qb, Kb);
  transpose_convert<<<dim3(256 / 32, 4096 / 32), blk, 0, stream>>>(Vf, VTb, 4096, 256);

  flash_attn<<<dim3(4096 / 128, 8), blk, 0, stream>>>(Qb, Kb, VTb, mask, AOb);

  gemm_bf16_async<<<dim3(2048 / 128, 4096 / 128), blk, 0, stream>>>(AOb, WoT, out, 4096, 2048, 2048);
}
